// GPT_49984829391250
// MI455X (gfx1250) — matrix operand order repaired and checked by reading, not compiled, not run on hardware
//
#include <hip/hip_runtime.h>
#include <math.h>

// ---------------------------------------------------------------------------
// Types for CDNA5 WMMA (gfx1250, wave32)
// ---------------------------------------------------------------------------
typedef __bf16 bhalf;
typedef bhalf v16bf __attribute__((ext_vector_type(16)));
typedef float v8f   __attribute__((ext_vector_type(8)));
typedef unsigned int u32x4 __attribute__((ext_vector_type(4)));
typedef int i32x4 __attribute__((ext_vector_type(4)));
typedef int i32x8 __attribute__((ext_vector_type(8)));

union FragAB { v16bf v; uint4 q[2]; };

// Tensor Data Mover availability (device pass only; host pass gets fallback)
#if defined(__HIP_DEVICE_COMPILE__) && __has_builtin(__builtin_amdgcn_tensor_load_to_lds)
#define HAVE_TDM 1
#else
#define HAVE_TDM 0
#endif

#if HAVE_TDM && __has_builtin(__builtin_amdgcn_s_wait_tensorcnt)
#define WAIT_TENSORCNT(n) __builtin_amdgcn_s_wait_tensorcnt(n)
#else
#define WAIT_TENSORCNT(n) ((void)0)
#endif

// Manual fp32 -> bf16 (round-to-nearest-even), stored as raw u16 bits.
__device__ __forceinline__ unsigned short f2bf_bits(float f) {
  unsigned u = __float_as_uint(f);
  u += 0x7FFFu + ((u >> 16) & 1u);
  return (unsigned short)(u >> 16);
}

// ---------------------------------------------------------------------------
// TDM: issue a 2D tile load (global -> LDS) per ISA cdna5 §8.3/8.4.
//   data_size = 2 bytes (bf16). LDS row padding applied by TDM:
//   pad_interval code: 2<<code dwords between pads; pad_amount code: code+1 dwords.
// ---------------------------------------------------------------------------
__device__ __forceinline__ void tdm_load_2d(
    unsigned lds_byte_off, unsigned long long gaddr,
    unsigned long long tensor_d0, unsigned long long tensor_d1,
    unsigned tile0, unsigned tile1,
    unsigned long long stride0,
    unsigned pad_interval, unsigned pad_amount) {
#if HAVE_TDM
  // ---- D# group 0 (128b): count=1, lds_addr, global_addr[56:0], type=2 ----
  u32x4 g0;
  g0[0] = 1u;                                   // count=1 (valid), user mode
  g0[1] = lds_byte_off;                         // lds_addr
  g0[2] = (unsigned)(gaddr & 0xFFFFFFFFull);    // global_addr[31:0]
  g0[3] = (unsigned)((gaddr >> 32) & 0x01FFFFFFull) | (2u << 30);  // type=2
  // ---- D# group 1 (256b) ----
  // q0 bits: [15:0] wg_mask=0 | [17:16] data_size=1(2B) | [20] pad_en
  //          [24:22] pad_interval | [31:25] pad_amount | [47:32] abar=0
  //          [63:48] tensor_dim0[15:0]
  unsigned long long q0 = (1ull << 16) | (1ull << 20)
                        | ((unsigned long long)(pad_interval & 7u) << 22)
                        | ((unsigned long long)(pad_amount & 127u) << 25)
                        | ((tensor_d0 & 0xFFFFull) << 48);
  // q1: [15:0] tensor_dim0[31:16] | [47:16] tensor_dim1 | [63:48] tile_dim0
  unsigned long long q1 = ((tensor_d0 >> 16) & 0xFFFFull)
                        | ((tensor_d1 & 0xFFFFFFFFull) << 16)
                        | ((unsigned long long)(tile0 & 0xFFFFu) << 48);
  // q2: [15:0] tile_dim1 | [31:16] tile_dim2=0 | [63:32] dim0_stride[31:0]
  unsigned long long q2 = (unsigned long long)(tile1 & 0xFFFFu)
                        | ((stride0 & 0xFFFFFFFFull) << 32);
  // q3: [15:0] dim0_stride[47:32] | [63:16] dim1_stride=0
  unsigned long long q3 = (stride0 >> 32) & 0xFFFFull;
  i32x8 g1;
  g1[0] = (int)(unsigned)(q0 & 0xFFFFFFFFull); g1[1] = (int)(unsigned)(q0 >> 32);
  g1[2] = (int)(unsigned)(q1 & 0xFFFFFFFFull); g1[3] = (int)(unsigned)(q1 >> 32);
  g1[4] = (int)(unsigned)(q2 & 0xFFFFFFFFull); g1[5] = (int)(unsigned)(q2 >> 32);
  g1[6] = (int)(unsigned)(q3 & 0xFFFFFFFFull); g1[7] = (int)(unsigned)(q3 >> 32);
  i32x4 z4 = {0, 0, 0, 0};
#if __clang_major__ >= 23
  i32x8 z8 = {0, 0, 0, 0, 0, 0, 0, 0};
  __builtin_amdgcn_tensor_load_to_lds(g0, g1, z4, z4, z8, 0);
#else
  __builtin_amdgcn_tensor_load_to_lds(g0, g1, z4, z4, 0);
#endif
#else
  (void)lds_byte_off; (void)gaddr; (void)tensor_d0; (void)tensor_d1;
  (void)tile0; (void)tile1; (void)stride0; (void)pad_interval; (void)pad_amount;
#endif
}

__device__ __forceinline__ unsigned lds_off_of(const void* p) {
  return (unsigned)(unsigned long long)p;  // low 32b of flat LDS address
}

// ---------------------------------------------------------------------------
// Block reductions (blockDim.x == 256 -> 8 wave32 waves)
// ---------------------------------------------------------------------------
__device__ __forceinline__ float warp_sum(float v) {
#pragma unroll
  for (int o = 16; o; o >>= 1) v += __shfl_xor(v, o, 32);
  return v;
}
__device__ __forceinline__ float warp_max(float v) {
#pragma unroll
  for (int o = 16; o; o >>= 1) v = fmaxf(v, __shfl_xor(v, o, 32));
  return v;
}
__device__ __forceinline__ float block_sum(float v, float* sb) {
  int tid = threadIdx.x;
  v = warp_sum(v);
  if ((tid & 31) == 0) sb[tid >> 5] = v;
  __syncthreads();
  float r = (tid < 8) ? sb[tid] : 0.0f;
  r = warp_sum(r);
  if (tid == 0) sb[0] = r;
  __syncthreads();
  r = sb[0];
  __syncthreads();
  return r;
}
__device__ __forceinline__ float block_max(float v, float* sb) {
  int tid = threadIdx.x;
  v = warp_max(v);
  if ((tid & 31) == 0) sb[tid >> 5] = v;
  __syncthreads();
  float r = (tid < 8) ? sb[tid] : -__builtin_inff();
  r = warp_max(r);
  if (tid == 0) sb[0] = r;
  __syncthreads();
  r = sb[0];
  __syncthreads();
  return r;
}

// ---------------------------------------------------------------------------
// fp32 -> bf16 weight conversion (streamed once per launch; ~free vs compute)
// ---------------------------------------------------------------------------
__global__ __launch_bounds__(256) void cvt_f32_bf16_kernel(
    const float* __restrict__ in, unsigned short* __restrict__ out, long long n) {
  long long i = (long long)blockIdx.x * blockDim.x + threadIdx.x;
  long long stride = (long long)gridDim.x * blockDim.x;
  for (; i < n; i += stride) out[i] = f2bf_bits(in[i]);
}

// ---------------------------------------------------------------------------
// Embedding: x[b,t,:] = tok_emb[idx[b,t],:] + pos_emb[t,:]
// ---------------------------------------------------------------------------
__global__ __launch_bounds__(256) void embed_kernel(
    const int* __restrict__ idx, const float* __restrict__ tok,
    const float* __restrict__ pos, float* __restrict__ x, int C, int T) {
  long long rt = blockIdx.x;
  int t = (int)(rt % T);
  int token = idx[rt];
  const float* te = tok + (long long)token * C;
  const float* pe = pos + (long long)t * C;
  float* xr = x + rt * (long long)C;
  for (int c = threadIdx.x; c < C; c += 256) xr[c] = te[c] + pe[c];
}

// ---------------------------------------------------------------------------
// LayerNorm (C == 1024, 256 threads, 4 elements/thread) -> bf16 output
// ---------------------------------------------------------------------------
__global__ __launch_bounds__(256) void layernorm_kernel(
    const float* __restrict__ x, const float* __restrict__ g,
    const float* __restrict__ b, unsigned short* __restrict__ out, int C) {
  __shared__ float sb[8];
  long long row = blockIdx.x;
  const float* xr = x + row * (long long)C;
  int tid = threadIdx.x;
  float vals[4];
  float s = 0.0f;
#pragma unroll
  for (int i = 0; i < 4; ++i) { vals[i] = xr[tid + i * 256]; s += vals[i]; }
  float mean = block_sum(s, sb) * (1.0f / 1024.0f);
  float vs = 0.0f;
#pragma unroll
  for (int i = 0; i < 4; ++i) { float d = vals[i] - mean; vs += d * d; }
  float var = block_sum(vs, sb) * (1.0f / 1024.0f);
  float rstd = rsqrtf(var + 1e-5f);
  unsigned short* orow = out + row * (long long)C;
#pragma unroll
  for (int i = 0; i < 4; ++i) {
    int c = tid + i * 256;
    orow[c] = f2bf_bits((vals[i] - mean) * rstd * g[c] + b[c]);
  }
}

// ---------------------------------------------------------------------------
// Causal softmax over rows of length T (== 1024), bf16 output
// ---------------------------------------------------------------------------
__global__ __launch_bounds__(256) void softmax_causal_kernel(
    const float* __restrict__ scores, unsigned short* __restrict__ att, int T) {
  __shared__ float sb[8];
  long long row = blockIdx.x;            // 0 .. B*T-1
  int t = (int)(row & (long long)(T - 1));
  const float* sr = scores + row * (long long)T;
  unsigned short* ar = att + row * (long long)T;
  int tid = threadIdx.x;
  float vals[4];
  float mx = -__builtin_inff();
#pragma unroll
  for (int i = 0; i < 4; ++i) {
    int sidx = tid + i * 256;
    float v = (sidx <= t) ? sr[sidx] : -__builtin_inff();
    vals[i] = v;
    mx = fmaxf(mx, v);
  }
  mx = block_max(mx, sb);
  float sum = 0.0f;
#pragma unroll
  for (int i = 0; i < 4; ++i) {
    int sidx = tid + i * 256;
    float e = (sidx <= t) ? expf(vals[i] - mx) : 0.0f;
    vals[i] = e;
    sum += e;
  }
  sum = block_sum(sum, sb);
  float inv = 1.0f / sum;
#pragma unroll
  for (int i = 0; i < 4; ++i) {
    int sidx = tid + i * 256;
    ar[sidx] = f2bf_bits(vals[i] * inv);
  }
}

// ---------------------------------------------------------------------------
// Shared WMMA tile helpers: block tile 128x128, K-tile 32, 8 waves,
// wave tile 32x64 (2x4 accumulators). Fragment layouts per ISA 7.12.2.
// ---------------------------------------------------------------------------
#define BM 128
#define BN 128
#define BKT 32
#define APAD 8
#define BPAD 8

__device__ __forceinline__ void wmma_compute_tile(
    const unsigned short (*Asb)[BKT + APAD],
    const unsigned short (*Bsb)[BN + BPAD],
    int wm, int wn, int lane, v8f acc[2][4]) {
  FragAB fa[2], fb[4];
  int arow  = lane & 15;
  int khalf = lane >> 4;  // 0: K=0..7,16..23   1: K=8..15,24..31
#pragma unroll
  for (int mi = 0; mi < 2; ++mi) {
    const unsigned short* p = &Asb[wm * 32 + mi * 16 + arow][khalf * 8];
    fa[mi].q[0] = *(const uint4*)(p);
    fa[mi].q[1] = *(const uint4*)(p + 16);
  }
#pragma unroll
  for (int nj = 0; nj < 4; ++nj) {
    const unsigned short* p = &Bsb[khalf * 8][wn * 64 + nj * 16 + arow];
    unsigned short* d = (unsigned short*)&fb[nj];
#pragma unroll
    for (int s = 0; s < 16; ++s)
      d[s] = p[(s < 8 ? s : s + 8) * (BN + BPAD)];
  }
#pragma unroll
  for (int mi = 0; mi < 2; ++mi)
#pragma unroll
    for (int nj = 0; nj < 4; ++nj)
      acc[mi][nj] = __builtin_amdgcn_wmma_f32_16x16x32_bf16(
          false, fa[mi].v, false, fb[nj].v, (short)0, acc[mi][nj],
          false, false);
}

__device__ __forceinline__ void wmma_epilogue(
    v8f acc[2][4], int wm, int wn, int lane,
    long long blockMo, long long blockNo, int N,
    const float* __restrict__ bias, const float* __restrict__ resb,
    float* __restrict__ outFb, unsigned short* __restrict__ outBb,
    float scale, int gelu) {
  int ncol = lane & 15;
  int m8   = (lane >> 4) * 8;
#pragma unroll
  for (int mi = 0; mi < 2; ++mi) {
#pragma unroll
    for (int nj = 0; nj < 4; ++nj) {
      long long ng = blockNo + wn * 64 + nj * 16 + ncol;
#pragma unroll
      for (int r = 0; r < 8; ++r) {
        long long mg = blockMo + wm * 32 + mi * 16 + m8 + r;
        float v = acc[mi][nj][r] * scale;
        if (bias) v += bias[ng];
        if (gelu) v = 0.5f * v * (1.0f + erff(v * 0.70710678118654752f));
        long long o = mg * (long long)N + ng;
        if (resb)  v += resb[o];
        if (outFb) outFb[o] = v;
        if (outBb) outBb[o] = f2bf_bits(v);
      }
    }
  }
}

// ---------------------------------------------------------------------------
// TDM-fed double-buffered WMMA GEMM (non-batched, B row-major [KxN]).
// Wave 0 drives the Tensor Data Mover: one D# per tile, TDM applies the
// LDS row padding; s_wait_tensorcnt + workgroup barrier release compute.
// Used for all weight GEMMs (QKV, MLP1/2, LM head).
// ---------------------------------------------------------------------------
__global__ __launch_bounds__(256) void wmma_gemm_bf16_tdm_kernel(
    const unsigned short* __restrict__ A, const unsigned short* __restrict__ B,
    int M, int N, int K,
    const float* __restrict__ bias, const float* __restrict__ residual,
    float* __restrict__ outF, unsigned short* __restrict__ outB, int gelu) {
  __shared__ __align__(16) unsigned short As[2][BM][BKT + APAD];
  __shared__ __align__(16) unsigned short Bs[2][BKT][BN + BPAD];

  const int tid  = threadIdx.x;
  const int lane = tid & 31;
  const int wid  = tid >> 5;
  const int wm   = wid >> 1;
  const int wn   = wid & 1;
  const long long blockMo = (long long)blockIdx.y * BM;
  const long long blockNo = (long long)blockIdx.x * BN;

  v8f acc[2][4] = {};
  const int nk = K / BKT;

#if HAVE_TDM
  if (wid == 0) {
    // Prologue: DMA first K-tile pair into buffer 0.
    tdm_load_2d(lds_off_of(&As[0][0][0]),
                (unsigned long long)(A + blockMo * (long long)K),
                (unsigned long long)K, (unsigned long long)M, BKT, BM,
                (unsigned long long)K, /*pad: every 16 dw*/ 3, /*+4 dw*/ 3);
    tdm_load_2d(lds_off_of(&Bs[0][0][0]),
                (unsigned long long)(B + blockNo),
                (unsigned long long)N, (unsigned long long)K, BN, BKT,
                (unsigned long long)N, /*pad: every 64 dw*/ 5, /*+4 dw*/ 3);
  }
#endif

  for (int kt = 0; kt < nk; ++kt) {
    const int cur = kt & 1;
#if HAVE_TDM
    if (wid == 0) {
      if (kt + 1 < nk) {
        const long long kk2 = (long long)(kt + 1) * BKT;
        tdm_load_2d(lds_off_of(&As[cur ^ 1][0][0]),
                    (unsigned long long)(A + blockMo * (long long)K + kk2),
                    (unsigned long long)K, (unsigned long long)M, BKT, BM,
                    (unsigned long long)K, 3, 3);
        tdm_load_2d(lds_off_of(&Bs[cur ^ 1][0][0]),
                    (unsigned long long)(B + kk2 * (long long)N + blockNo),
                    (unsigned long long)N, (unsigned long long)K, BN, BKT,
                    (unsigned long long)N, 5, 3);
        WAIT_TENSORCNT(2);   // current pair resident; next pair in flight
      } else {
        WAIT_TENSORCNT(0);
      }
    }
#else
    {  // fallback: manual staging by all threads
      const int kk = kt * BKT;
#pragma unroll
      for (int it = 0; it < 2; ++it) {
        int idx = tid + it * 256;
        int row = idx >> 2, c4 = idx & 3;
        *(uint4*)&As[cur][row][c4 * 8] =
            *(const uint4*)(A + (blockMo + row) * (long long)K + kk + c4 * 8);
      }
#pragma unroll
      for (int it = 0; it < 2; ++it) {
        int idx = tid + it * 256;
        int row = idx >> 4, c8 = idx & 15;
        *(uint4*)&Bs[cur][row][c8 * 8] =
            *(const uint4*)(B + (long long)(kk + row) * N + blockNo + c8 * 8);
      }
    }
#endif
    __syncthreads();
    wmma_compute_tile(As[cur], Bs[cur], wm, wn, lane, acc);
    __syncthreads();  // all waves done reading before TDM refills this buffer
  }

  wmma_epilogue(acc, wm, wn, lane, blockMo, blockNo, N,
                bias, residual, outF, outB, 1.0f, gelu);
}

// ---------------------------------------------------------------------------
// Batched WMMA GEMM (attention): optional transB ([NxK] gather) + scale,
// residual fusion, f32/bf16 outputs. Manual LDS staging + L2 prefetch.
// ---------------------------------------------------------------------------
__global__ __launch_bounds__(256) void wmma_gemm_bf16_kernel(
    const unsigned short* __restrict__ A, const unsigned short* __restrict__ B,
    int M, int N, int K,
    long long strideA, long long strideB, long long strideO,
    const float* __restrict__ bias,
    const float* __restrict__ residual, long long strideR,
    float* __restrict__ outF, unsigned short* __restrict__ outB,
    float scale, int transB, int gelu) {
  __shared__ __align__(16) unsigned short As[BM][BKT + APAD];
  __shared__ __align__(16) unsigned short Bs[BKT][BN + BPAD];

  const int tid  = threadIdx.x;
  const int lane = tid & 31;
  const int wid  = tid >> 5;
  const int wm   = wid >> 1;
  const int wn   = wid & 1;
  const int z    = blockIdx.z;
  const long long blockMo = (long long)blockIdx.y * BM;
  const long long blockNo = (long long)blockIdx.x * BN;

  const unsigned short* Ab = A + (long long)z * strideA;
  const unsigned short* Bb = B + (long long)z * strideB;

  v8f acc[2][4] = {};

  for (int kk = 0; kk < K; kk += BKT) {
#pragma unroll
    for (int it = 0; it < 2; ++it) {
      int idx = tid + it * 256;
      int row = idx >> 2, c4 = idx & 3;
      *(uint4*)&As[row][c4 * 8] =
          *(const uint4*)(Ab + (blockMo + row) * (long long)K + kk + c4 * 8);
    }
    if (!transB) {
#pragma unroll
      for (int it = 0; it < 2; ++it) {
        int idx = tid + it * 256;
        int row = idx >> 4, c8 = idx & 15;
        *(uint4*)&Bs[row][c8 * 8] =
            *(const uint4*)(Bb + (long long)(kk + row) * N + blockNo + c8 * 8);
      }
    } else {  // B global is [N x K]: Bs[k][n] = Bg[(blockN+n)*K + kk + k]
#pragma unroll
      for (int it = 0; it < 2; ++it) {
        int idx = tid + it * 256;
        int n = idx >> 2, k4 = idx & 3;
        union { uint4 q; unsigned short u[8]; } tmp;
        tmp.q = *(const uint4*)(Bb + (blockNo + n) * (long long)K + kk + k4 * 8);
#pragma unroll
        for (int j = 0; j < 8; ++j) Bs[k4 * 8 + j][n] = tmp.u[j];
      }
    }
    if (kk + BKT < K) {  // L2 prefetch of next tile (global_prefetch_b8)
      __builtin_prefetch(Ab + (blockMo + (tid >> 1)) * (long long)K + kk + BKT, 0, 1);
      if (!transB)
        __builtin_prefetch(Bb + (long long)(kk + BKT + (tid >> 4)) * N + blockNo, 0, 1);
      else
        __builtin_prefetch(Bb + (blockNo + (tid >> 1)) * (long long)K + kk + BKT, 0, 1);
    }
    __syncthreads();
    wmma_compute_tile(As, Bs, wm, wn, lane, acc);
    __syncthreads();
  }

  float* outFb = outF ? outF + (long long)z * strideO : (float*)0;
  unsigned short* outBb = outB ? outB + (long long)z * strideO : (unsigned short*)0;
  const float* resb = residual ? residual + (long long)z * strideR : (const float*)0;
  wmma_epilogue(acc, wm, wn, lane, blockMo, blockNo, N,
                bias, resb, outFb, outBb, scale, gelu);
}

// ---------------------------------------------------------------------------
// Loss: per-row log-softmax NLL over V logits, then deterministic reduction
// ---------------------------------------------------------------------------
__global__ __launch_bounds__(256) void loss_row_kernel(
    const float* __restrict__ logits, const int* __restrict__ targets,
    float* __restrict__ row_loss, int Vv) {
  __shared__ float sb[8];
  long long row = blockIdx.x;
  const float* lr = logits + row * (long long)Vv;
  float mx = -__builtin_inff();
  for (int i = threadIdx.x; i < Vv; i += 256) mx = fmaxf(mx, lr[i]);
  mx = block_max(mx, sb);
  float s = 0.0f;
  for (int i = threadIdx.x; i < Vv; i += 256) s += expf(lr[i] - mx);
  s = block_sum(s, sb);
  if (threadIdx.x == 0) {
    int tgt = targets[row];
    row_loss[row] = -(lr[tgt] - mx - logf(s));
  }
}

__global__ __launch_bounds__(256) void loss_reduce_kernel(
    const float* __restrict__ row_loss, float* __restrict__ out, int n) {
  __shared__ float sb[8];
  float s = 0.0f;
  for (int i = threadIdx.x; i < n; i += 256) s += row_loss[i];
  s = block_sum(s, sb);
  if (threadIdx.x == 0) out[0] = s / (float)n;
}

// ---------------------------------------------------------------------------
// Host orchestration
// ---------------------------------------------------------------------------
extern "C" void kernel_launch(void* const* d_in, const int* in_sizes, int n_in,
                              void* d_out, int out_size, void* d_ws,
                              size_t ws_size, hipStream_t stream) {
  (void)in_sizes; (void)n_in; (void)out_size; (void)ws_size;

  constexpr int L = 8, C = 1024, V = 32000, B = 4, T = 1024, H = 4 * C;
  constexpr int M = B * T;  // 4096 token rows

  const int*   idx     = (const int*)d_in[0];
  const int*   targets = (const int*)d_in[1];
  const float* tok_emb = (const float*)d_in[2];
  const float* pos_emb = (const float*)d_in[3];
  const float* ln1_g   = (const float*)d_in[4];
  const float* ln1_b   = (const float*)d_in[5];
  const float* Wq      = (const float*)d_in[6];
  const float* bq      = (const float*)d_in[7];
  const float* Wk      = (const float*)d_in[8];
  const float* bk      = (const float*)d_in[9];
  const float* Wv      = (const float*)d_in[10];
  const float* bv      = (const float*)d_in[11];
  const float* ln2_g   = (const float*)d_in[12];
  const float* ln2_b   = (const float*)d_in[13];
  const float* W1      = (const float*)d_in[14];
  const float* b1      = (const float*)d_in[15];
  const float* W2      = (const float*)d_in[16];
  const float* b2      = (const float*)d_in[17];
  const float* lnf_g   = (const float*)d_in[18];
  const float* lnf_b   = (const float*)d_in[19];
  const float* Wh      = (const float*)d_in[20];
  const float* bh      = (const float*)d_in[21];

  char* wsp = (char*)d_ws;
  auto alloc = [&](size_t bytes) -> void* {
    void* p = wsp;
    wsp += (bytes + 255) & ~(size_t)255;
    return p;
  };
  unsigned short* wq_bf = (unsigned short*)alloc((size_t)L * C * C * 2);
  unsigned short* wk_bf = (unsigned short*)alloc((size_t)L * C * C * 2);
  unsigned short* wv_bf = (unsigned short*)alloc((size_t)L * C * C * 2);
  unsigned short* w1_bf = (unsigned short*)alloc((size_t)L * C * H * 2);
  unsigned short* w2_bf = (unsigned short*)alloc((size_t)L * H * C * 2);
  unsigned short* wh_bf = (unsigned short*)alloc((size_t)C * V * 2);
  float*          x     = (float*)alloc((size_t)M * C * 4);
  unsigned short* h_bf  = (unsigned short*)alloc((size_t)M * C * 2);
  unsigned short* q_bf  = (unsigned short*)alloc((size_t)M * C * 2);
  unsigned short* k_bf  = (unsigned short*)alloc((size_t)M * C * 2);
  unsigned short* v_bf  = (unsigned short*)alloc((size_t)M * C * 2);
  unsigned short* h1_bf = (unsigned short*)alloc((size_t)M * H * 2);
  float*          scores = (float*)alloc((size_t)B * T * T * 4);
  unsigned short* att_bf = (unsigned short*)alloc((size_t)B * T * T * 2);
  float*          row_loss = (float*)alloc((size_t)M * 4);

  auto cvt = [&](const float* s, unsigned short* d, long long n) {
    cvt_f32_bf16_kernel<<<8192, 256, 0, stream>>>(s, d, n);
  };
  cvt(Wq, wq_bf, (long long)L * C * C);
  cvt(Wk, wk_bf, (long long)L * C * C);
  cvt(Wv, wv_bf, (long long)L * C * C);
  cvt(W1, w1_bf, (long long)L * C * H);
  cvt(W2, w2_bf, (long long)L * H * C);
  cvt(Wh, wh_bf, (long long)C * V);

  // Weight GEMMs: TDM-fed double-buffered kernel.
  auto gemm_w = [&](const unsigned short* Am, const unsigned short* Bm,
                    int Mm, int Nn, int Kk, const float* bias,
                    const float* residual, float* outF, unsigned short* outB,
                    int gelu) {
    dim3 grid(Nn / 128, Mm / 128, 1);
    wmma_gemm_bf16_tdm_kernel<<<grid, 256, 0, stream>>>(
        Am, Bm, Mm, Nn, Kk, bias, residual, outF, outB, gelu);
  };
  // Attention GEMMs: batched kernel (transB / scale / residual).
  auto gemm_a = [&](const unsigned short* Am, const unsigned short* Bm,
                    int Mm, int Nn, int Kk,
                    long long sA, long long sB, long long sO, int batch,
                    const float* residual, long long sR,
                    float* outF, unsigned short* outB,
                    float scale, int transB) {
    dim3 grid(Nn / 128, Mm / 128, batch);
    wmma_gemm_bf16_kernel<<<grid, 256, 0, stream>>>(
        Am, Bm, Mm, Nn, Kk, sA, sB, sO, nullptr, residual, sR, outF, outB,
        scale, transB, 0);
  };

  embed_kernel<<<M, 256, 0, stream>>>(idx, tok_emb, pos_emb, x, C, T);

  const long long TC = (long long)T * C;
  const long long TT = (long long)T * T;

  for (int l = 0; l < L; ++l) {
    layernorm_kernel<<<M, 256, 0, stream>>>(x, ln1_g + l * C, ln1_b + l * C,
                                            h_bf, C);
    gemm_w(h_bf, wq_bf + (long long)l * C * C, M, C, C, bq + l * C,
           nullptr, nullptr, q_bf, 0);
    gemm_w(h_bf, wk_bf + (long long)l * C * C, M, C, C, bk + l * C,
           nullptr, nullptr, k_bf, 0);
    gemm_w(h_bf, wv_bf + (long long)l * C * C, M, C, C, bv + l * C,
           nullptr, nullptr, v_bf, 0);
    // scores = (Q @ K^T) / sqrt(C)   [batched, transB]
    gemm_a(q_bf, k_bf, T, T, C, TC, TC, TT, B,
           nullptr, 0, scores, nullptr, 0.03125f, 1);
    softmax_causal_kernel<<<M, 256, 0, stream>>>(scores, att_bf, T);
    // x = x + att @ V   [batched, residual fused]
    gemm_a(att_bf, v_bf, T, C, T, TT, TC, TC, B,
           x, TC, x, nullptr, 1.0f, 0);
    layernorm_kernel<<<M, 256, 0, stream>>>(x, ln2_g + l * C, ln2_b + l * C,
                                            h_bf, C);
    gemm_w(h_bf, w1_bf + (long long)l * C * H, M, H, C, b1 + l * H,
           nullptr, nullptr, h1_bf, 1);
    gemm_w(h1_bf, w2_bf + (long long)l * H * C, M, C, H, b2 + l * C,
           x, x, nullptr, 0);
  }

  layernorm_kernel<<<M, 256, 0, stream>>>(x, lnf_g, lnf_b, h_bf, C);
  float* logits = (float*)d_out;
  gemm_w(h_bf, wh_bf, M, V, C, bh, nullptr, logits, nullptr, 0);

  loss_row_kernel<<<M, 256, 0, stream>>>(logits, targets, row_loss, V);
  loss_reduce_kernel<<<1, 256, 0, stream>>>(row_loss,
                                            logits + (long long)M * V, M);
}
